// ImageCaptionModel_44590350467455
// MI455X (gfx1250) — matrix operand order repaired and checked by reading, not compiled, not run on hardware
//
#include <hip/hip_runtime.h>
#include <hip/hip_bf16.h>
#include <math.h>

// Model dims
#define BB 32
#define TT 40
#define VV 32000
#define DE 512
#define HH 512
#define DC 2048

typedef __bf16 bf16;
typedef float  v8f   __attribute__((ext_vector_type(8)));
typedef float  v2f   __attribute__((ext_vector_type(2)));
typedef bf16   v16bf __attribute__((ext_vector_type(16)));
typedef bf16   v8bf  __attribute__((ext_vector_type(8)));
typedef unsigned int u32x4 __attribute__((ext_vector_type(4)));
typedef int    i32x8 __attribute__((ext_vector_type(8)));
typedef int    i32x4 __attribute__((ext_vector_type(4)));

// ---------------------------------------------------------------------------
// Prep: convert W_out [V, H] fp32 -> bf16 (L2-resident operand for logits GEMM)
// ---------------------------------------------------------------------------
__global__ void convert_wout_kernel(const float* __restrict__ src,
                                    bf16* __restrict__ dst) {
  size_t i = ((size_t)blockIdx.x * blockDim.x + threadIdx.x) * 8;
#pragma unroll
  for (int j = 0; j < 8; ++j) dst[i + j] = (bf16)src[i + j];
}

__global__ void zero_kernel(float* __restrict__ ptr, int n) {
  int i = blockIdx.x * blockDim.x + threadIdx.x;
  if (i < n) ptr[i] = 0.0f;
}

// ---------------------------------------------------------------------------
// Input layer: p = leaky_relu(cnn @ W_in^T + b_in), [32,2048]x[2048,512]
// ---------------------------------------------------------------------------
__global__ void input_layer_kernel(const float* __restrict__ cnn,
                                   const float* __restrict__ W_in,
                                   const float* __restrict__ b_in,
                                   float* __restrict__ p) {
  int b = blockIdx.x;
  int h = threadIdx.x;
  const float* x = cnn + (size_t)b * DC;
  const float* w = W_in + (size_t)h * DC;
  float acc = 0.0f;
#pragma unroll 8
  for (int k = 0; k < DC; ++k) acc = fmaf(x[k], w[k], acc);
  acc += b_in[h];
  p[b * HH + h] = acc > 0.0f ? acc : 0.01f * acc;
}

// ---------------------------------------------------------------------------
// One RNN step: h_new = tanh([e_t | p | h_prev] @ W_rnn + b_rnn)
//   M=32, K=1536, N=512 in exact fp32 via V_WMMA_F32_16X16X4_F32.
// The concat is virtual: A fragments are gathered from emb/p/h_prev by region.
// ---------------------------------------------------------------------------
__global__ void rnn_step_kernel(const float* __restrict__ emb,
                                const int*   __restrict__ xtok,
                                const float* __restrict__ p,
                                const float* __restrict__ Wr,   // [1536,512]
                                const float* __restrict__ br,   // [512]
                                const float* __restrict__ hprev,// [32,512]
                                float* __restrict__ hout,       // [32,512]
                                bf16*  __restrict__ hbf,        // [T,32,512]
                                int t) {
  const int wave  = threadIdx.x >> 5;
  const int lane  = threadIdx.x & 31;
  const int tile  = blockIdx.x * 4 + wave;   // 0..63
  const int mBase = (tile & 1) * 16;
  const int nBase = (tile >> 1) * 16;
  const int half  = lane >> 4;
  const int nl    = lane & 15;
  const int m     = mBase + nl;
  const int n     = nBase + nl;
  const int koff  = half * 2;

  const int tok = xtok[m * TT + t];
  const float* aRows[3];
  aRows[0] = emb + (size_t)tok * DE;
  aRows[1] = p + (size_t)m * HH;
  aRows[2] = hprev + (size_t)m * HH;

  v8f acc = {};
#pragma unroll
  for (int region = 0; region < 3; ++region) {
    const float* aRow  = aRows[region];
    const float* wBase = Wr + (size_t)(region * 512) * HH + n;
#pragma unroll 4
    for (int kk = 0; kk < 512; kk += 4) {
      const float2 av = *(const float2*)(aRow + kk + koff);
      v2f a; a.x = av.x; a.y = av.y;
      const float* wp = wBase + (size_t)(kk + koff) * HH;
      v2f bv; bv.x = wp[0]; bv.y = wp[HH];
      acc = __builtin_amdgcn_wmma_f32_16x16x4_f32(false, a, false, bv,
                                                  (short)0, acc, false, false);
    }
  }

  const float bias = br[n];
#pragma unroll
  for (int r = 0; r < 8; ++r) {
    const int mm = mBase + half * 8 + r;
    const float hv = tanhf(acc[r] + bias);
    hout[mm * HH + n] = hv;
    hbf[((size_t)t * BB + mm) * HH + n] = (bf16)hv;
  }
}

// ---------------------------------------------------------------------------
// Fused logits GEMM: [1280,512]bf16 x [512,32000]bf16 -> f32, K=512.
// Block = 256 threads (8 waves) covering 64 M x 128 N.
// Wave = 2Mx2N register tile (4 accumulators) -> A/B fragments reused 2x each.
// Per K-step the block's B tile (128 vocab cols x 32 K bf16 = 8 KB) is pulled
// into LDS by the Tensor Data Mover (one tensor_load_to_lds, 2D D#), completed
// with s_wait_tensorcnt and published with the workgroup barrier; B fragments
// are then ds_load_b128 reads shared by all 8 waves. This cuts W_out L2
// traffic ~4x vs per-wave global fragment loads.
// ---------------------------------------------------------------------------
__global__ void logits_kernel(const bf16* __restrict__ Hbf,   // [T,32,512]
                              const bf16* __restrict__ Wb,    // [V,512] bf16
                              const float* __restrict__ b_out,
                              float* __restrict__ out) {      // [B,T,V]
  __shared__ bf16 smemB[128 * 32];   // [col][k] row-major per col

  const int wave  = threadIdx.x >> 5;   // 0..7
  const int lane  = threadIdx.x & 31;
  const int waveM = wave & 1;           // 2 M wave-groups
  const int waveN = wave >> 1;          // 4 N wave-groups
  const int half  = lane >> 4;
  const int nl    = lane & 15;

  const int mBlock = blockIdx.y * 64;
  const int nBlock = blockIdx.x * 128;

  int mBase0 = mBlock + waveM * 32;
  int mBase1 = mBase0 + 16;
  int nLoc0  = waveN * 32;              // local col within block tile
  int nLoc1  = nLoc0 + 16;

  // A row pointers: output row r = b*T + t -> hidden row t*B + b
  const bf16* aRow0;
  const bf16* aRow1;
  {
    int r0 = mBase0 + nl, b0 = r0 / TT, t0 = r0 - b0 * TT;
    int r1 = mBase1 + nl, b1 = r1 / TT, t1 = r1 - b1 * TT;
    aRow0 = Hbf + ((size_t)t0 * BB + b0) * HH;
    aRow1 = Hbf + ((size_t)t1 * BB + b1) * HH;
  }

  const unsigned lds_base =
      (unsigned)(unsigned long long)&smemB[0];            // LDS byte offset
  const unsigned long long wb_base =
      (unsigned long long)(Wb + (size_t)nBlock * HH);     // tile origin (bytes)

  v8f acc00 = {}, acc01 = {}, acc10 = {}, acc11 = {};

  for (int k0 = 0; k0 < HH; k0 += 32) {
    __syncthreads();   // previous B tile fully consumed before overwrite
    if (wave == 0) {
#if __has_builtin(__builtin_amdgcn_tensor_load_to_lds)
      // D# per CDNA5 ISA:
      //   2D tile: tile_dim0=32 (K, contiguous bf16), tile_dim1=128 (vocab
      //   rows), tensor_dim0_stride=512, data_size=2B, no pad/iterate/mcast.
      const unsigned long long ga = wb_base + (unsigned)(k0 * 2);
      u32x4 g0; i32x8 g1; i32x4 g2 = {}; i32x4 g3 = {}; i32x8 g4 = {};
      g0.x = 1u;                                            // count=1
      g0.y = lds_base;                                      // lds_addr
      g0.z = (unsigned)ga;                                  // global_addr lo
      g0.w = (unsigned)((ga >> 32) & 0x01FFFFFFu)           // global_addr hi
             | 0x80000000u;                                 // type=2 (image)
      g1[0] = (int)(1u << 16);      // data_size=1 -> 2 bytes
      g1[1] = (int)(32u << 16);     // tensor_dim0[15:0] in bits 63:48
      g1[2] = (int)(128u << 16);    // tensor_dim0 hi=0 | tensor_dim1 lo=128
      g1[3] = (int)(32u << 16);     // tensor_dim1 hi=0 | tile_dim0=32
      g1[4] = 128;                  // tile_dim1=128 | tile_dim2=0
      g1[5] = 512;                  // tensor_dim0_stride lo32 = 512
      g1[6] = 0;                    // stride0 hi | stride1 lo
      g1[7] = 0;                    // stride1 hi
      // amdgpu-toolchain (clang-23) 6-arg form: extra i32x8 group + cpol
      __builtin_amdgcn_tensor_load_to_lds(g0, g1, g2, g3, g4, 0);
      __builtin_amdgcn_s_wait_tensorcnt(0);
#else
      // Fallback: cooperative copy by wave 0 (4096 bf16 = 512 x v8bf)
      const v8bf* src = (const v8bf*)(Wb + (size_t)nBlock * HH + k0);
      for (int i = lane; i < 128 * 32 / 8; i += 32) {
        int col = i / 4, part = i & 3;   // 4 x v8bf per 32-K col row
        ((v8bf*)smemB)[i] =
            *(const v8bf*)(Wb + ((size_t)nBlock + col) * HH + k0 + part * 8);
      }
      (void)src;
#endif
    }
    __syncthreads();   // B tile visible to all waves

    // A fragments from global bf16 (16-bit A 16x32 lane layout)
    union { v16bf v; v8bf h[2]; } A0, A1, Bm0, Bm1;
    A0.h[0] = *(const v8bf*)(aRow0 + k0 + half * 8);
    A0.h[1] = *(const v8bf*)(aRow0 + k0 + 16 + half * 8);
    A1.h[0] = *(const v8bf*)(aRow1 + k0 + half * 8);
    A1.h[1] = *(const v8bf*)(aRow1 + k0 + 16 + half * 8);
    // B fragments from LDS: lane col = nLoc + nl, K = 8*half+{0..7}, 16+8*half+{0..7}
    const bf16* b0 = &smemB[(nLoc0 + nl) * 32 + half * 8];
    const bf16* b1 = &smemB[(nLoc1 + nl) * 32 + half * 8];
    Bm0.h[0] = *(const v8bf*)(b0);
    Bm0.h[1] = *(const v8bf*)(b0 + 16);
    Bm1.h[0] = *(const v8bf*)(b1);
    Bm1.h[1] = *(const v8bf*)(b1 + 16);

    acc00 = __builtin_amdgcn_wmma_f32_16x16x32_bf16(false, A0.v, false, Bm0.v,
                                                    (short)0, acc00, false, false);
    acc01 = __builtin_amdgcn_wmma_f32_16x16x32_bf16(false, A0.v, false, Bm1.v,
                                                    (short)0, acc01, false, false);
    acc10 = __builtin_amdgcn_wmma_f32_16x16x32_bf16(false, A1.v, false, Bm0.v,
                                                    (short)0, acc10, false, false);
    acc11 = __builtin_amdgcn_wmma_f32_16x16x32_bf16(false, A1.v, false, Bm1.v,
                                                    (short)0, acc11, false, false);
  }

  // Epilogue: bias + store, C/D layout (VGPR r -> row, lane half -> +8)
  const int n0 = nBlock + nLoc0 + nl;
  const int n1 = nBlock + nLoc1 + nl;
  const float bias0 = b_out[n0];
  const float bias1 = b_out[n1];
#pragma unroll
  for (int rr = 0; rr < 8; ++rr) {
    const int m0 = mBase0 + half * 8 + rr;
    const int m1 = mBase1 + half * 8 + rr;
    out[(size_t)m0 * VV + n0] = acc00[rr] + bias0;
    out[(size_t)m0 * VV + n1] = acc01[rr] + bias1;
    out[(size_t)m1 * VV + n0] = acc10[rr] + bias0;
    out[(size_t)m1 * VV + n1] = acc11[rr] + bias1;
  }
}

// ---------------------------------------------------------------------------
// Host launcher
// ---------------------------------------------------------------------------
extern "C" void kernel_launch(void* const* d_in, const int* in_sizes, int n_in,
                              void* d_out, int out_size, void* d_ws, size_t ws_size,
                              hipStream_t stream) {
  const float* cnn   = (const float*)d_in[0];
  const int*   xtok  = (const int*)  d_in[1];
  // d_in[2] = is_train -> identity in deterministic reference, ignored
  const float* W_in  = (const float*)d_in[3];
  const float* b_in  = (const float*)d_in[4];
  const float* emb   = (const float*)d_in[5];
  const float* W_rnn = (const float*)d_in[6];
  const float* b_rnn = (const float*)d_in[7];
  const float* W_out = (const float*)d_in[8];
  const float* b_out = (const float*)d_in[9];
  float* out = (float*)d_out;

  // Workspace layout (16B aligned)
  char* ws = (char*)d_ws;
  bf16*  WoutB = (bf16*)ws;                                    // 32,768,000 B
  float* Hfp   = (float*)(ws + 32768000);                      // (T+1)*32*512*4
  bf16*  Hbf   = (bf16*)(ws + 32768000 + 2686976);             // T*32*512*2
  float* p     = (float*)(ws + 32768000 + 2686976 + 1310720);  // 32*512*4

  convert_wout_kernel<<<8000, 256, 0, stream>>>(W_out, WoutB);
  zero_kernel<<<(BB * HH + 255) / 256, 256, 0, stream>>>(Hfp, BB * HH);
  input_layer_kernel<<<BB, HH, 0, stream>>>(cnn, W_in, b_in, p);

  for (int t = 0; t < TT; ++t) {
    rnn_step_kernel<<<16, 128, 0, stream>>>(emb, xtok, p, W_rnn, b_rnn,
                                            Hfp + (size_t)t * BB * HH,
                                            Hfp + (size_t)(t + 1) * BB * HH,
                                            Hbf, t);
  }

  logits_kernel<<<dim3(VV / 128, (BB * TT) / 64), 256, 0, stream>>>(
      Hbf, WoutB, b_out, out);

  (void)hipMemcpyAsync(out + (size_t)BB * TT * VV, Hfp + (size_t)TT * BB * HH,
                       (size_t)BB * HH * sizeof(float), hipMemcpyDeviceToDevice,
                       stream);
}
